// E3GNN_35416300323524
// MI455X (gfx1250) — matrix operand order repaired and checked by reading, not compiled, not run on hardware
//
#include <hip/hip_runtime.h>
#include <hip/hip_bf16.h>
#include <math.h>

typedef __attribute__((ext_vector_type(16))) _Float16 v16h;
typedef __attribute__((ext_vector_type(8)))  _Float16 v8h;
typedef __attribute__((ext_vector_type(8)))  float    v8f;

#define HID      256
#define NNODES   10000
#define NEDGES   200000
#define NGRAPHS  500
#define EIN      581
#define EIN_PAD  608      /* 19 * 32 */
#define KB_E1    19
#define KB_256   8
#define KB_512   16
#define NT       16       /* 256 / 16 output n-tiles */
#define EM       64       /* edges per block (M tile) */
#define ASTR_E   616      /* LDS stride (halves) for edge A tile  */
#define ASTR_N   520      /* LDS stride (halves) for node A tile  */
#define HSTR     264      /* LDS stride (halves) for hidden tile  */

__device__ __forceinline__ float silu_f(float x) {
    return x / (1.0f + __expf(-x));
}

// ---------------------------------------------------------------------------
// Weight pre-swizzle: pack W[K,256] (fp32) into WMMA-B fragment order (f16).
// Fragment entry index = ((kb*16 + nt)*32 + lane)*16 + i
//   lane holds column n = nt*16 + (lane&15),
//   k rows = kb*32 + (lane>>4)*16 + i      (i = 0..15)
// ---------------------------------------------------------------------------
__global__ void k_swizzle_w(const float* __restrict__ W, _Float16* __restrict__ out,
                            int Krows, int KB) {
    int idx = blockIdx.x * blockDim.x + threadIdx.x;
    int total = KB * NT * 32 * 16;
    if (idx >= total) return;
    int i    =  idx        & 15;
    int lane = (idx >> 4)  & 31;
    int nt   = (idx >> 9)  & 15;
    int kb   =  idx >> 13;
    int k = kb * 32 + ((lane >> 4) << 4) + i;
    int n = (nt << 4) + (lane & 15);
    float v = (k < Krows) ? W[k * HID + n] : 0.0f;
    out[idx] = (_Float16)v;
}

// lattice inner products: ips[g][i][k] = sum_j L[g][i][j] * L[g][k][j]
__global__ void k_latips(const float* __restrict__ L, float* __restrict__ out) {
    int idx = blockIdx.x * blockDim.x + threadIdx.x;
    if (idx >= NGRAPHS * 9) return;
    int g = idx / 9;
    int r = idx - g * 9;
    int i = r / 3, k = r - (r / 3) * 3;
    const float* Lg = L + g * 9;
    out[idx] = Lg[i*3+0]*Lg[k*3+0] + Lg[i*3+1]*Lg[k*3+1] + Lg[i*3+2]*Lg[k*3+2];
}

__global__ void k_zero(float* __restrict__ p, int n) {
    int i = blockIdx.x * blockDim.x + threadIdx.x;
    if (i < n) p[i] = 0.0f;
}

__global__ void k_count(const int* __restrict__ src, float* __restrict__ cnt) {
    int e = blockIdx.x * blockDim.x + threadIdx.x;
    if (e < NEDGES) atomicAdd(&cnt[src[e]], 1.0f);
}

// ---------------------------------------------------------------------------
// WMMA GEMM helper, 8 waves per block, 2 n-tiles per wave, MS M-subtiles.
// Per k-step: load 2 B fragments once, reuse across MS A fragments
// -> 2*MS WMMAs per k-step, B L2-traffic amortized over MS*16 rows.
// ---------------------------------------------------------------------------
template<int MS, int KB>
__device__ __forceinline__ void wmma_gemm(const _Float16* __restrict__ Ash,
                                          int astride,
                                          const _Float16* __restrict__ Bsw,
                                          int lane, int wave, v8f (&acc)[MS][2]) {
    const int arow  = lane & 15;
    const int koff8 = (lane >> 4) * 16;
    for (int kb = 0; kb < KB; ++kb) {
        const _Float16* bb = Bsw + (size_t)((kb * NT + wave * 2) * 32 + lane) * 16;
        v16h b0 = *(const v16h*)(bb);
        v16h b1 = *(const v16h*)(bb + 32 * 16);
#pragma unroll
        for (int s = 0; s < MS; ++s) {
            v16h a;
            const _Float16* ap = Ash + (s * 16 + arow) * astride + kb * 32 + koff8;
            ((v8h*)&a)[0] = *(const v8h*)(ap);        // K + {0..7}
            ((v8h*)&a)[1] = *(const v8h*)(ap + 8);    // K + {8..15}
            acc[s][0] = __builtin_amdgcn_wmma_f32_16x16x32_f16(
                false, a, false, b0, (short)0, acc[s][0], false, false);
            acc[s][1] = __builtin_amdgcn_wmma_f32_16x16x32_f16(
                false, a, false, b1, (short)0, acc[s][1], false, false);
        }
    }
}

template<int MS>
__device__ __forceinline__ void zero_acc(v8f (&acc)[MS][2]) {
#pragma unroll
    for (int s = 0; s < MS; ++s) {
        v8f z = {};
        acc[s][0] = z;
        acc[s][1] = z;
    }
}

// ---------------------------------------------------------------------------
// Edge kernel: 64 edges per block, 256 threads (8 waves).
// Gather e_in -> LDS f16, GEMM1 (K=608), SiLU -> LDS, GEMM2 (K=256), SiLU,
// atomic scatter-add into agg[src].
// ---------------------------------------------------------------------------
__global__ __launch_bounds__(256, 1)
void k_edge(const float* __restrict__ nf, const float* __restrict__ frac,
            const int* __restrict__ src, const int* __restrict__ dst,
            const int* __restrict__ e2g, const float* __restrict__ latips,
            const _Float16* __restrict__ We1s, const float* __restrict__ be1,
            const _Float16* __restrict__ We2s, const float* __restrict__ be2,
            float* __restrict__ agg) {
    __shared__ __align__(32) _Float16 Ash[EM * ASTR_E];   // 78848 B
    __shared__ __align__(32) _Float16 Hsh[EM * HSTR];     // 33792 B

    const int tid   = threadIdx.x;
    const int ebase = blockIdx.x * EM;

    // ---- Phase 1: build e_in tile [64 x 608] in LDS (f16) ----
    for (int i = tid; i < EM * EIN_PAD; i += 256) {
        int row = i / EIN_PAD;
        int col = i - row * EIN_PAD;
        int e   = ebase + row;
        float v;
        if (col < 256) {
            v = nf[src[e] * HID + col];
        } else if (col < 512) {
            v = nf[dst[e] * HID + (col - 256)];
        } else if (col < 521) {
            v = latips[e2g[e] * 9 + (col - 512)];
        } else if (col < 581) {
            int c    = col - 521;      // 0..59
            int trig = c / 30;         // 0 = sin, 1 = cos
            int r30  = c - trig * 30;
            int d    = r30 / 10;
            int f    = r30 - d * 10;
            float fd = frac[dst[e] * 3 + d] - frac[src[e] * 3 + d];
            fd = fd - floorf(fd);      // mod 1.0
            float ang = 6.28318530717958647692f * (float)f * fd;
            v = trig ? __cosf(ang) : __sinf(ang);
        } else {
            v = 0.0f;                  // K padding 581..607
        }
        Ash[row * ASTR_E + col] = (_Float16)v;
    }
    __syncthreads();

    const int lane  = tid & 31;
    const int wave  = tid >> 5;           // 0..7 -> n-tiles 2w, 2w+1
    const int mbase = (lane >> 4) << 3;   // C/D rows: v + mbase (+ 16*s)
    const int ncol  = lane & 15;

    // ---- Phase 2: GEMM1 [64x608] x [608x256] ----
    v8f acc[4][2];
    zero_acc(acc);
    wmma_gemm<4, KB_E1>(Ash, ASTR_E, We1s, lane, wave, acc);

    // ---- Phase 3: bias + SiLU -> Hsh (f16) ----
#pragma unroll
    for (int s = 0; s < 4; ++s) {
#pragma unroll
        for (int t = 0; t < 2; ++t) {
            int n    = (wave * 2 + t) * 16 + ncol;
            float bv = be1[n];
#pragma unroll
            for (int v = 0; v < 8; ++v) {
                float x = acc[s][t][v] + bv;
                Hsh[(s * 16 + mbase + v) * HSTR + n] = (_Float16)silu_f(x);
            }
        }
    }
    __syncthreads();

    // ---- Phase 4: GEMM2 [64x256] x [256x256] ----
    zero_acc(acc);
    wmma_gemm<4, KB_256>(Hsh, HSTR, We2s, lane, wave, acc);

    // ---- Phase 5: bias + SiLU + atomic scatter into agg[src] ----
#pragma unroll
    for (int s = 0; s < 4; ++s) {
        int srcs[8];
#pragma unroll
        for (int v = 0; v < 8; ++v) srcs[v] = src[ebase + s * 16 + mbase + v];
#pragma unroll
        for (int t = 0; t < 2; ++t) {
            int n    = (wave * 2 + t) * 16 + ncol;
            float bv = be2[n];
#pragma unroll
            for (int v = 0; v < 8; ++v) {
                float sv = silu_f(acc[s][t][v] + bv);
                atomicAdd(&agg[srcs[v] * HID + n], sv);
            }
        }
    }
}

// ---------------------------------------------------------------------------
// Node kernel: 16 nodes per block, 256 threads (8 waves, 2 n-tiles each).
// n_in = [nf | agg/max(cnt,1)] -> GEMM (K=512) -> SiLU -> GEMM (K=256)
// -> SiLU -> out = nf + h.
// ---------------------------------------------------------------------------
__global__ __launch_bounds__(256, 1)
void k_node(const float* __restrict__ nf, const float* __restrict__ agg,
            const float* __restrict__ cnt,
            const _Float16* __restrict__ Wn1s, const float* __restrict__ bn1,
            const _Float16* __restrict__ Wn2s, const float* __restrict__ bn2,
            float* __restrict__ out) {
    __shared__ __align__(32) _Float16 Ash[16 * ASTR_N];
    __shared__ __align__(32) _Float16 Hsh[16 * HSTR];

    const int tid   = threadIdx.x;
    const int nbase = blockIdx.x * 16;

    // ---- Phase 1: build n_in tile [16 x 512] in LDS (f16) ----
    for (int i = tid; i < 16 * 512; i += 256) {
        int row  = i >> 9;
        int col  = i & 511;
        int node = nbase + row;
        float v;
        if (col < 256) {
            v = nf[node * HID + col];
        } else {
            float c = fmaxf(cnt[node], 1.0f);
            v = agg[node * HID + (col - 256)] / c;
        }
        Ash[row * ASTR_N + col] = (_Float16)v;
    }
    __syncthreads();

    const int lane  = tid & 31;
    const int wave  = tid >> 5;
    const int mbase = (lane >> 4) << 3;
    const int ncol  = lane & 15;

    // ---- GEMM1 [16x512] x [512x256] ----
    v8f acc[1][2];
    zero_acc(acc);
    wmma_gemm<1, KB_512>(Ash, ASTR_N, Wn1s, lane, wave, acc);

#pragma unroll
    for (int t = 0; t < 2; ++t) {
        int n    = (wave * 2 + t) * 16 + ncol;
        float bv = bn1[n];
#pragma unroll
        for (int v = 0; v < 8; ++v) {
            float x = acc[0][t][v] + bv;
            Hsh[(mbase + v) * HSTR + n] = (_Float16)silu_f(x);
        }
    }
    __syncthreads();

    // ---- GEMM2 [16x256] x [256x256] ----
    zero_acc(acc);
    wmma_gemm<1, KB_256>(Hsh, HSTR, Wn2s, lane, wave, acc);

    // ---- residual + store ----
#pragma unroll
    for (int t = 0; t < 2; ++t) {
        int n    = (wave * 2 + t) * 16 + ncol;
        float bv = bn2[n];
#pragma unroll
        for (int v = 0; v < 8; ++v) {
            int node = nbase + mbase + v;
            float s  = silu_f(acc[0][t][v] + bv);
            out[node * HID + n] = nf[node * HID + n] + s;
        }
    }
}

// ---------------------------------------------------------------------------
// Host launch
// ---------------------------------------------------------------------------
extern "C" void kernel_launch(void* const* d_in, const int* in_sizes, int n_in,
                              void* d_out, int out_size, void* d_ws, size_t ws_size,
                              hipStream_t stream) {
    const float* nf   = (const float*)d_in[0];
    const float* frac = (const float*)d_in[1];
    const float* lat  = (const float*)d_in[2];
    const int*   ei   = (const int*)d_in[3];   // [2, E] : src row 0, dst row 1
    const int*   e2g  = (const int*)d_in[4];
    const float* We1  = (const float*)d_in[5];
    const float* be1  = (const float*)d_in[6];
    const float* We2  = (const float*)d_in[7];
    const float* be2  = (const float*)d_in[8];
    const float* Wn1  = (const float*)d_in[9];
    const float* bn1  = (const float*)d_in[10];
    const float* Wn2  = (const float*)d_in[11];
    const float* bn2  = (const float*)d_in[12];
    float* out = (float*)d_out;

    const int* src = ei;
    const int* dst = ei + NEDGES;

    // Workspace layout (bytes)
    constexpr size_t SZ_WE1  = (size_t)KB_E1  * NT * 32 * 16 * 2;   // 311296
    constexpr size_t SZ_W256 = (size_t)KB_256 * NT * 32 * 16 * 2;   // 131072
    constexpr size_t SZ_WN1  = (size_t)KB_512 * NT * 32 * 16 * 2;   // 262144
    constexpr size_t OFF_WE1 = 0;
    constexpr size_t OFF_WE2 = OFF_WE1 + SZ_WE1;
    constexpr size_t OFF_WN1 = OFF_WE2 + SZ_W256;
    constexpr size_t OFF_WN2 = OFF_WN1 + SZ_WN1;
    constexpr size_t OFF_LAT = OFF_WN2 + SZ_W256;
    constexpr size_t OFF_CNT = OFF_LAT + (size_t)NGRAPHS * 9 * 4;
    constexpr size_t OFF_AGG = OFF_CNT + (size_t)NNODES * 4;        // contiguous after cnt

    char* ws = (char*)d_ws;
    _Float16* We1s  = (_Float16*)(ws + OFF_WE1);
    _Float16* We2s  = (_Float16*)(ws + OFF_WE2);
    _Float16* Wn1s  = (_Float16*)(ws + OFF_WN1);
    _Float16* Wn2s  = (_Float16*)(ws + OFF_WN2);
    float*    latip = (float*)(ws + OFF_LAT);
    float*    cnt   = (float*)(ws + OFF_CNT);
    float*    agg   = (float*)(ws + OFF_AGG);

    // --- prep: weight swizzle (f32 -> WMMA-fragment f16) ---
    {
        int t1 = KB_E1  * NT * 32 * 16;
        int t2 = KB_256 * NT * 32 * 16;
        int t3 = KB_512 * NT * 32 * 16;
        k_swizzle_w<<<(t1 + 255) / 256, 256, 0, stream>>>(We1, We1s, EIN, KB_E1);
        k_swizzle_w<<<(t2 + 255) / 256, 256, 0, stream>>>(We2, We2s, HID, KB_256);
        k_swizzle_w<<<(t3 + 255) / 256, 256, 0, stream>>>(Wn1, Wn1s, 2 * HID, KB_512);
        k_swizzle_w<<<(t2 + 255) / 256, 256, 0, stream>>>(Wn2, Wn2s, HID, KB_256);
    }
    // --- prep: lattice inner products ---
    k_latips<<<(NGRAPHS * 9 + 255) / 256, 256, 0, stream>>>(lat, latip);

    // --- zero cnt + agg (contiguous) ---
    {
        int nz = NNODES + NNODES * HID;
        k_zero<<<(nz + 255) / 256, 256, 0, stream>>>(cnt, nz);
    }
    // --- degree counts ---
    k_count<<<(NEDGES + 255) / 256, 256, 0, stream>>>(src, cnt);

    // --- edge MLP + scatter (64 edges / block) ---
    k_edge<<<NEDGES / EM, 256, 0, stream>>>(nf, frac, src, dst, e2g, latip,
                                            We1s, be1, We2s, be2, agg);

    // --- node MLP + residual ---
    k_node<<<NNODES / 16, 256, 0, stream>>>(nf, agg, cnt, Wn1s, bn1, Wn2s, bn2, out);

    (void)in_sizes; (void)n_in; (void)out_size; (void)ws_size;
}
